// LatentAttentionBlock_30562987279067
// MI455X (gfx1250) — matrix operand order repaired and checked by reading, not compiled, not run on hardware
//
#include <hip/hip_runtime.h>
#include <hip/hip_bf16.h>
#include <math.h>

// ---- problem constants -----------------------------------------------------
#define NB   128          // batch
#define SEQ  200          // sequence length
#define SP   224          // sequence padded to multiple of 32
#define DIM  1024         // model dim D
#define QKD  128          // q/k dim
#define EXP  2048         // E = 2*D
#define NTOK (NB*SEQ)     // 25600 tokens
#define WE_ROWS 4352      // 2*QK + 2*E rows of expand_w

typedef __attribute__((ext_vector_type(16))) __bf16 v16bf;
typedef __attribute__((ext_vector_type(8)))  float  v8f;

union FragU { v16bf v; uint4 q[2]; };

static __device__ __forceinline__ v8f zero8() {
  v8f z;
#pragma unroll
  for (int i = 0; i < 8; ++i) z[i] = 0.0f;
  return z;
}

// B-operand frag: 16 contiguous bf16 (32 bytes) at p
static __device__ __forceinline__ v16bf ld_b_frag(const __bf16* p) {
  FragU f;
  f.q[0] = ((const uint4*)p)[0];
  f.q[1] = ((const uint4*)p)[1];
  return f.v;
}
// A-operand frag: 16 contiguous bf16 at p
static __device__ __forceinline__ v16bf ld_a_frag(const __bf16* p) {
  FragU f;
  f.q[0] = *(const uint4*)(p);
  f.q[1] = *(const uint4*)(p + 8);
  return f.v;
}

static __device__ __forceinline__ v8f wmma_bf16(v16bf a, v16bf b, v8f c) {
  return __builtin_amdgcn_wmma_f32_16x16x32_bf16(false, a, false, b, (short)0, c,
                                                 false, false);
}

static __device__ __forceinline__ float gelu_exact(float x) {
  return 0.5f * x * (1.0f + erff(x * 0.7071067811865476f));
}

// ---- Tensor Data Mover staging (6-arg builtin on this toolchain) -----------
#if __has_builtin(__builtin_amdgcn_tensor_load_to_lds) && \
    __has_builtin(__builtin_amdgcn_s_wait_tensorcnt)
#define HAVE_TDM 1
typedef __attribute__((ext_vector_type(4))) unsigned int u32x4;
typedef __attribute__((ext_vector_type(8))) int i32x8;
typedef __attribute__((ext_vector_type(4))) int i32x4;

// Load a 2D tile (tile_rows x tile_k bf16, row stride = stride_elems) from
// global into LDS at byte offset lds_off, packed contiguously row-major.
static __device__ __forceinline__ void
tdm_stage_b(unsigned lds_off, const __bf16* gsrc, unsigned tile_k,
            unsigned tile_rows, unsigned long long stride_elems) {
  unsigned long long ga = (unsigned long long)(uintptr_t)gsrc;
  u32x4 g0;
  g0[0] = 1u;                                           // count=1, user mode
  g0[1] = lds_off;                                      // lds_addr (bytes)
  g0[2] = (unsigned)ga;                                 // global addr lo
  g0[3] = (unsigned)((ga >> 32) & 0x01FFFFFFu) | (2u << 30);  // hi + type=2
  unsigned td0 = (unsigned)stride_elems;                // tensor dim0
  unsigned td1 = 1u << 20;                              // tensor dim1 (ample)
  i32x8 g1;
  g1[0] = 1 << 16;                                      // data_size = 2 bytes
  g1[1] = (int)((td0 & 0xFFFFu) << 16);                 // dim0 lo16
  g1[2] = (int)(((td0 >> 16) & 0xFFFFu) | ((td1 & 0xFFFFu) << 16));
  g1[3] = (int)(((td1 >> 16) & 0xFFFFu) | (tile_k << 16));   // tile_dim0
  g1[4] = (int)(tile_rows & 0xFFFFu);                   // tile_dim1, tile_dim2=0
  g1[5] = (int)(unsigned)(stride_elems & 0xFFFFFFFFu);  // dim0 stride lo32
  g1[6] = (int)(unsigned)((stride_elems >> 32) & 0xFFFFu);   // stride hi16
  g1[7] = 0;
  i32x4 z4 = {0, 0, 0, 0};
  i32x8 z8 = {0, 0, 0, 0, 0, 0, 0, 0};
  __builtin_amdgcn_tensor_load_to_lds(g0, g1, z4, z4, z8, 0);
}
#endif

// ---- kernel 1: LayerNorm (f32 in) -> bf16 xn -------------------------------
__global__ void __launch_bounds__(256)
ln_kernel(const float* __restrict__ x, const float* __restrict__ g,
          const float* __restrict__ be, __bf16* __restrict__ xn) {
  const int row = blockIdx.x;
  const float* xr = x + (size_t)row * DIM;
  float s1 = 0.f, s2 = 0.f;
  for (int j = threadIdx.x; j < DIM; j += 256) {
    float v = xr[j];
    s1 += v; s2 += v * v;
  }
#pragma unroll
  for (int d = 16; d > 0; d >>= 1) {
    s1 += __shfl_xor(s1, d, 32);
    s2 += __shfl_xor(s2, d, 32);
  }
  __shared__ float sh1[8], sh2[8], mb[2];
  const int wv = threadIdx.x >> 5, ln = threadIdx.x & 31;
  if (ln == 0) { sh1[wv] = s1; sh2[wv] = s2; }
  __syncthreads();
  if (threadIdx.x == 0) {
    float a = 0.f, b2 = 0.f;
    for (int i = 0; i < 8; ++i) { a += sh1[i]; b2 += sh2[i]; }
    float mean = a * (1.0f / DIM);
    float var  = b2 * (1.0f / DIM) - mean * mean;
    mb[0] = mean;
    mb[1] = rsqrtf(var + 1e-5f);
  }
  __syncthreads();
  const float mean = mb[0], rstd = mb[1];
  for (int j = threadIdx.x; j < DIM; j += 256)
    xn[(size_t)row * DIM + j] = (__bf16)((xr[j] - mean) * rstd * g[j] + be[j]);
}

// ---- kernel 2: f32 -> bf16 cast -------------------------------------------
__global__ void __launch_bounds__(256)
cvt_kernel(const float* __restrict__ src, __bf16* __restrict__ dst, int n) {
  int i = blockIdx.x * 256 + threadIdx.x;
  if (i < n) dst[i] = (__bf16)src[i];
}

// ---- kernel 3: zero padded key rows of gvT --------------------------------
__global__ void __launch_bounds__(256)
gvt_pad_kernel(__bf16* __restrict__ gvT) {
  int idx = blockIdx.x * 256 + threadIdx.x;
  const int pad = SP - SEQ;
  const int total = NB * DIM * pad;
  if (idx >= total) return;
  int s = SEQ + idx % pad;
  int r = idx / pad;                       // r = b*DIM + v
  gvT[(size_t)r * SP + s] = (__bf16)0.0f;
}

// ---- kernel 4: h[:, 0:256] = xn @ expand_w[0:256]^T  (q|k) ----------------
// B tile (64 cols x 32 K) staged in LDS (TDM double-buffered when available),
// shared by all 8 waves; A register-double-buffered from global.
__global__ void __launch_bounds__(256)
gemm_qk_kernel(const __bf16* __restrict__ xn, const __bf16* __restrict__ wE,
               __bf16* __restrict__ hqk) {
  __shared__ __bf16 bsm[2][64 * 32];       // 2 x 4KB, at LDS offset 0
  const int l = threadIdx.x & 31, wv = threadIdx.x >> 5;
  const int mbase = blockIdx.x * 128 + wv * 16;
  const int cbase = blockIdx.y * 64;
  const int KS = DIM / 32;
  const __bf16* arow = xn + (size_t)(mbase + (l & 15)) * DIM + ((l >> 4) << 4);
  const __bf16* bsrc = wE + (size_t)cbase * DIM;

  // prologue: stage K-step 0 into buffer 0
#ifdef HAVE_TDM
  if (threadIdx.x < 32) {
    tdm_stage_b(0u, bsrc, 32u, 64u, (unsigned long long)DIM);
    __builtin_amdgcn_s_wait_tensorcnt(0);
  }
#else
  {
    int t = threadIdx.x;
    if (t < 128) {
      int e = t * 8, r = e >> 5, c = e & 31;
      *(uint4*)(&bsm[0][e]) = *(const uint4*)(bsrc + (size_t)r * DIM + c);
    }
  }
#endif

  v16bf ac = ld_a_frag(arow);
  v8f acc[4];
#pragma unroll
  for (int s = 0; s < 4; ++s) acc[s] = zero8();

  for (int kk = 0; kk < KS; ++kk) {
    __syncthreads();                       // staged data visible, next buf free
    const int nb = (kk + 1) & 1;
#ifdef HAVE_TDM
    if (threadIdx.x < 32 && kk + 1 < KS)
      tdm_stage_b((unsigned)(nb * 64 * 32 * 2), bsrc + (kk + 1) * 32, 32u, 64u,
                  (unsigned long long)DIM);
#else
    if (kk + 1 < KS) {
      int t = threadIdx.x;
      if (t < 128) {
        int e = t * 8, r = e >> 5, c = e & 31;
        *(uint4*)(&bsm[nb][e]) =
            *(const uint4*)(bsrc + (size_t)r * DIM + (kk + 1) * 32 + c);
      }
    }
#endif
    const int o = (kk + 1 < KS) ? (kk + 1) * 32 : 0;
    v16bf an = ld_a_frag(arow + o);        // prefetch next A frag
    const __bf16* bt = bsm[kk & 1];
#pragma unroll
    for (int s = 0; s < 4; ++s) {
      v16bf bf = ld_b_frag(bt + (size_t)(s * 16 + (l & 15)) * 32 + ((l >> 4) << 4));
      acc[s] = wmma_bf16(ac, bf, acc[s]);
    }
#ifdef HAVE_TDM
    if (threadIdx.x < 32 && kk + 1 < KS) __builtin_amdgcn_s_wait_tensorcnt(0);
#endif
    ac = an;
  }
#pragma unroll
  for (int s = 0; s < 4; ++s)
#pragma unroll
    for (int i = 0; i < 8; ++i) {
      int n = mbase + i + ((l >> 4) << 3);
      int c = cbase + s * 16 + (l & 15);
      hqk[(size_t)n * 256 + c] = (__bf16)acc[s][i];
    }
}

// ---- kernel 5: fused GEGLU: lin*gelu(pre); local -> cat, value -> gvT^T ---
__global__ void __launch_bounds__(256)
gemm_geglu_kernel(const __bf16* __restrict__ xn, const __bf16* __restrict__ wE,
                  __bf16* __restrict__ cat, __bf16* __restrict__ gvT) {
  const int l = threadIdx.x & 31, wv = threadIdx.x >> 5;
  const int mbase = blockIdx.x * 128 + wv * 16;
  const int j0 = blockIdx.y * 64;
  const int KS = DIM / 32;
  const __bf16* arow = xn + (size_t)(mbase + (l & 15)) * DIM + ((l >> 4) << 4);
  const __bf16* blin = wE + (size_t)(256  + j0 + (l & 15)) * DIM + ((l >> 4) << 4);
  const __bf16* bpre = wE + (size_t)(2304 + j0 + (l & 15)) * DIM + ((l >> 4) << 4);
  v8f aL[4], aP[4];
#pragma unroll
  for (int s = 0; s < 4; ++s) { aL[s] = zero8(); aP[s] = zero8(); }

  v16bf ac = ld_a_frag(arow);
  v16bf blc[4], bpc[4];
#pragma unroll
  for (int s = 0; s < 4; ++s) {
    blc[s] = ld_b_frag(blin + (size_t)(s * 16) * DIM);
    bpc[s] = ld_b_frag(bpre + (size_t)(s * 16) * DIM);
  }
  for (int kk = 0; kk < KS; ++kk) {
    const int o = (kk + 1 < KS) ? (kk + 1) * 32 : 0;
    __builtin_prefetch(arow + o + 256, 0, 0);        // L2 prefetch ahead
    v16bf an = ld_a_frag(arow + o);
    v16bf bln[4], bpn[4];
#pragma unroll
    for (int s = 0; s < 4; ++s) {
      bln[s] = ld_b_frag(blin + (size_t)(s * 16) * DIM + o);
      bpn[s] = ld_b_frag(bpre + (size_t)(s * 16) * DIM + o);
    }
#pragma unroll
    for (int s = 0; s < 4; ++s) {
      aL[s] = wmma_bf16(ac, blc[s], aL[s]);
      aP[s] = wmma_bf16(ac, bpc[s], aP[s]);
    }
    ac = an;
#pragma unroll
    for (int s = 0; s < 4; ++s) { blc[s] = bln[s]; bpc[s] = bpn[s]; }
  }
#pragma unroll
  for (int s = 0; s < 4; ++s)
#pragma unroll
    for (int i = 0; i < 8; ++i) {
      int n = mbase + i + ((l >> 4) << 3);
      int j = j0 + s * 16 + (l & 15);
      float gval = aL[s][i] * gelu_exact(aP[s][i]);
      if (j < DIM) {                              // geglu_local
        cat[(size_t)n * EXP + j] = (__bf16)gval;
      } else {                                    // geglu_value (transposed)
        int v = j - DIM;
        int b = n / SEQ, sq = n - b * SEQ;
        gvT[((size_t)b * DIM + v) * SP + sq] = (__bf16)gval;
      }
    }
}

// ---- kernel 6: per-batch scores = q k^T * scale + bias; softmax -> attn ---
__global__ void __launch_bounds__(32)
attn_softmax_kernel(const __bf16* __restrict__ hqk, const float* __restrict__ pbm_in,
                    __bf16* __restrict__ attn) {
  const int l = threadIdx.x;
  const int b = blockIdx.x, qt = blockIdx.y;
  const float p = pbm_in[0];
  const float sp = (p > 20.f) ? p : log1pf(__expf(p));   // softplus
  const float scale = 0.08838834764831845f;              // 1/sqrt(128)

  int sq = qt * 16 + (l & 15); if (sq > SEQ - 1) sq = SEQ - 1;   // clamp pad rows
  const __bf16* qrow = hqk + (size_t)(b * SEQ + sq) * 256 + ((l >> 4) << 4);
  v16bf qf[4];
#pragma unroll
  for (int kk = 0; kk < 4; ++kk) qf[kk] = ld_a_frag(qrow + kk * 32);

  v8f acc[SP / 16];
  v16bf bc[4];
  {
    int sk = (l & 15); if (sk > SEQ - 1) sk = SEQ - 1;
    const __bf16* krow = hqk + (size_t)(b * SEQ + sk) * 256 + 128 + ((l >> 4) << 4);
#pragma unroll
    for (int kk = 0; kk < 4; ++kk) bc[kk] = ld_b_frag(krow + kk * 32);
  }
#pragma unroll
  for (int kt = 0; kt < SP / 16; ++kt) {
    int ktn = (kt + 1 < SP / 16) ? kt + 1 : kt;
    int sk = ktn * 16 + (l & 15); if (sk > SEQ - 1) sk = SEQ - 1;
    const __bf16* krow = hqk + (size_t)(b * SEQ + sk) * 256 + 128 + ((l >> 4) << 4);
    v16bf bn[4];
#pragma unroll
    for (int kk = 0; kk < 4; ++kk) bn[kk] = ld_b_frag(krow + kk * 32);
    v8f a = zero8();
#pragma unroll
    for (int kk = 0; kk < 4; ++kk) a = wmma_bf16(qf[kk], bc[kk], a);
    acc[kt] = a;
#pragma unroll
    for (int kk = 0; kk < 4; ++kk) bc[kk] = bn[kk];
  }
  // causal mask + linear relative-position bias
#pragma unroll
  for (int kt = 0; kt < SP / 16; ++kt)
#pragma unroll
    for (int i = 0; i < 8; ++i) {
      int m = qt * 16 + i + ((l >> 4) << 3);
      int n = kt * 16 + (l & 15);
      float v = acc[kt][i];
      acc[kt][i] = (n <= m && n < SEQ) ? (v * scale + sp * (float)(n - m)) : -3.0e38f;
    }
  // row softmax (keys spread over the 16 lanes of each half-wave)
#pragma unroll
  for (int i = 0; i < 8; ++i) {
    float mx = -3.0e38f;
#pragma unroll
    for (int kt = 0; kt < SP / 16; ++kt) mx = fmaxf(mx, acc[kt][i]);
#pragma unroll
    for (int d = 1; d < 16; d <<= 1) mx = fmaxf(mx, __shfl_xor(mx, d, 32));
    float sum = 0.f;
#pragma unroll
    for (int kt = 0; kt < SP / 16; ++kt) {
      float e = __expf(acc[kt][i] - mx);
      acc[kt][i] = e; sum += e;
    }
#pragma unroll
    for (int d = 1; d < 16; d <<= 1) sum += __shfl_xor(sum, d, 32);
    float inv = 1.0f / sum;
    int m = qt * 16 + i + ((l >> 4) << 3);
#pragma unroll
    for (int kt = 0; kt < SP / 16; ++kt)
      attn[((size_t)b * SP + m) * SP + kt * 16 + (l & 15)] = (__bf16)(acc[kt][i] * inv);
  }
}

// ---- kernel 7: att_out = attn @ geglu_value -> cat[:, 1024:2048] ----------
__global__ void __launch_bounds__(128)
gemm_pv_kernel(const __bf16* __restrict__ attn, const __bf16* __restrict__ gvT,
               __bf16* __restrict__ cat) {
  const int l = threadIdx.x & 31, wv = threadIdx.x >> 5;
  const int b = blockIdx.x, qt = blockIdx.y;
  const int cbase = blockIdx.z * 256 + wv * 64;
  const int KS = SP / 32;
  const __bf16* arow = attn + ((size_t)b * SP + qt * 16 + (l & 15)) * SP + ((l >> 4) << 4);
  const __bf16* brow = gvT + ((size_t)b * DIM + cbase + (l & 15)) * SP + ((l >> 4) << 4);
  v8f acc[4];
#pragma unroll
  for (int s = 0; s < 4; ++s) acc[s] = zero8();
  v16bf ac = ld_a_frag(arow);
  v16bf bc[4];
#pragma unroll
  for (int s = 0; s < 4; ++s) bc[s] = ld_b_frag(brow + (size_t)(s * 16) * SP);
#pragma unroll
  for (int kk = 0; kk < KS; ++kk) {
    const int o = (kk + 1 < KS) ? (kk + 1) * 32 : 0;
    v16bf an = ld_a_frag(arow + o);
    v16bf bn[4];
#pragma unroll
    for (int s = 0; s < 4; ++s) bn[s] = ld_b_frag(brow + (size_t)(s * 16) * SP + o);
#pragma unroll
    for (int s = 0; s < 4; ++s) acc[s] = wmma_bf16(ac, bc[s], acc[s]);
    ac = an;
#pragma unroll
    for (int s = 0; s < 4; ++s) bc[s] = bn[s];
  }
#pragma unroll
  for (int s = 0; s < 4; ++s)
#pragma unroll
    for (int i = 0; i < 8; ++i) {
      int sq = qt * 16 + i + ((l >> 4) << 3);
      if (sq < SEQ) {
        int c = cbase + s * 16 + (l & 15);
        cat[((size_t)(b * SEQ + sq)) * EXP + DIM + c] = (__bf16)acc[s][i];
      }
    }
}

// ---- kernel 8: out = cat @ project_w^T + x (f32), 32 rows per wave --------
__global__ void __launch_bounds__(256)
gemm_proj_kernel(const __bf16* __restrict__ cat, const __bf16* __restrict__ wP,
                 const float* __restrict__ x, float* __restrict__ out) {
  const int l = threadIdx.x & 31, wv = threadIdx.x >> 5;
  const int mbase = blockIdx.x * 256 + wv * 32;      // 32 rows per wave
  const int cbase = blockIdx.y * 64;
  const int KS = EXP / 32;
  const __bf16* arow0 = cat + (size_t)(mbase + (l & 15)) * EXP + ((l >> 4) << 4);
  const __bf16* arow1 = arow0 + (size_t)16 * EXP;
  const __bf16* brow  = wP + (size_t)(cbase + (l & 15)) * EXP + ((l >> 4) << 4);
  v8f acc0[4], acc1[4];
#pragma unroll
  for (int s = 0; s < 4; ++s) { acc0[s] = zero8(); acc1[s] = zero8(); }

  v16bf a0c = ld_a_frag(arow0), a1c = ld_a_frag(arow1);
  v16bf bc[4];
#pragma unroll
  for (int s = 0; s < 4; ++s) bc[s] = ld_b_frag(brow + (size_t)(s * 16) * EXP);
  for (int kk = 0; kk < KS; ++kk) {
    const int o = (kk + 1 < KS) ? (kk + 1) * 32 : 0;
    __builtin_prefetch(arow0 + o + 256, 0, 0);
    __builtin_prefetch(arow1 + o + 256, 0, 0);
    v16bf a0n = ld_a_frag(arow0 + o), a1n = ld_a_frag(arow1 + o);
    v16bf bn[4];
#pragma unroll
    for (int s = 0; s < 4; ++s) bn[s] = ld_b_frag(brow + (size_t)(s * 16) * EXP + o);
#pragma unroll
    for (int s = 0; s < 4; ++s) {
      acc0[s] = wmma_bf16(a0c, bc[s], acc0[s]);
      acc1[s] = wmma_bf16(a1c, bc[s], acc1[s]);
    }
    a0c = a0n; a1c = a1n;
#pragma unroll
    for (int s = 0; s < 4; ++s) bc[s] = bn[s];
  }
#pragma unroll
  for (int s = 0; s < 4; ++s)
#pragma unroll
    for (int i = 0; i < 8; ++i) {
      int n0 = mbase + i + ((l >> 4) << 3);
      int c  = cbase + s * 16 + (l & 15);
      out[(size_t)n0 * DIM + c] = acc0[s][i] + x[(size_t)n0 * DIM + c];
      int n1 = n0 + 16;
      out[(size_t)n1 * DIM + c] = acc1[s][i] + x[(size_t)n1 * DIM + c];
    }
}

// ---- host launch -----------------------------------------------------------
static constexpr size_t aln(size_t v) { return (v + 255) & ~(size_t)255; }

extern "C" void kernel_launch(void* const* d_in, const int* in_sizes, int n_in,
                              void* d_out, int out_size, void* d_ws, size_t ws_size,
                              hipStream_t stream) {
  (void)in_sizes; (void)n_in; (void)out_size; (void)ws_size;
  const float* x   = (const float*)d_in[0];
  const float* eW  = (const float*)d_in[1];
  const float* pW  = (const float*)d_in[2];
  const float* gam = (const float*)d_in[3];
  const float* bet = (const float*)d_in[4];
  const float* pbm = (const float*)d_in[5];
  float* out = (float*)d_out;

  const size_t SZ_XN  = (size_t)NTOK * DIM * 2;
  const size_t SZ_WE  = (size_t)WE_ROWS * DIM * 2;
  const size_t SZ_WP  = (size_t)DIM * EXP * 2;
  const size_t SZ_HQK = (size_t)NTOK * 256 * 2;
  const size_t SZ_CAT = (size_t)NTOK * EXP * 2;
  const size_t SZ_GVT = (size_t)NB * DIM * SP * 2;

  char* base = (char*)d_ws;
  size_t off = 0;
  __bf16* xn  = (__bf16*)(base + off); off = aln(off + SZ_XN);
  __bf16* wE  = (__bf16*)(base + off); off = aln(off + SZ_WE);
  __bf16* wP  = (__bf16*)(base + off); off = aln(off + SZ_WP);
  __bf16* hqk = (__bf16*)(base + off); off = aln(off + SZ_HQK);
  __bf16* cat = (__bf16*)(base + off); off = aln(off + SZ_CAT);
  __bf16* gvT = (__bf16*)(base + off); off = aln(off + SZ_GVT);
  __bf16* att = (__bf16*)(base + off);

  ln_kernel<<<NTOK, 256, 0, stream>>>(x, gam, bet, xn);
  cvt_kernel<<<(WE_ROWS * DIM + 255) / 256, 256, 0, stream>>>(eW, wE, WE_ROWS * DIM);
  cvt_kernel<<<(DIM * EXP + 255) / 256, 256, 0, stream>>>(pW, wP, DIM * EXP);
  gvt_pad_kernel<<<(NB * DIM * (SP - SEQ) + 255) / 256, 256, 0, stream>>>(gvT);

  gemm_qk_kernel<<<dim3(NTOK / 128, 4), 256, 0, stream>>>(xn, wE, hqk);
  gemm_geglu_kernel<<<dim3(NTOK / 128, EXP / 64), 256, 0, stream>>>(xn, wE, cat, gvT);
  attn_softmax_kernel<<<dim3(NB, SP / 16), 32, 0, stream>>>(hqk, pbm, att);
  gemm_pv_kernel<<<dim3(NB, SP / 16, 4), 128, 0, stream>>>(att, gvT, cat);
  gemm_proj_kernel<<<dim3(NTOK / 256, DIM / 64), 256, 0, stream>>>(cat, wP, x, out);
}
